// KANLayer_85512798864151
// MI455X (gfx1250) — matrix operand order repaired and checked by reading, not compiled, not run on hardware
//
#include <hip/hip_runtime.h>

typedef __attribute__((ext_vector_type(16))) _Float16 v16h;
typedef __attribute__((ext_vector_type(8)))  _Float16 v8h;
typedef __attribute__((ext_vector_type(4)))  _Float16 v4h;
typedef __attribute__((ext_vector_type(8)))  float    v8f;
typedef __attribute__((ext_vector_type(4)))  float    v4f;
typedef __attribute__((ext_vector_type(4)))  int      v4i;

#define BATCH 4096
#define INF   1024
#define OUTF  1024
#define KCHEB 8
#define KDIM  (INF * KCHEB)   // 8192 = GEMM K dimension

#define BM 128
#define BN 128
#define BK 64                 // K-halves staged per LDS buffer
#define ASTRIDE (BK + 8)      // padded LDS row stride (halves) to spread banks

#if __has_builtin(__builtin_amdgcn_global_load_async_to_lds_b128) && \
    __has_builtin(__builtin_amdgcn_s_wait_asynccnt)
#define USE_ASYNC_LDS 1
// Builtin params are pointers-to-v4i in the global (AS1) / LDS (AS3) spaces.
typedef __attribute__((address_space(1))) v4i g_v4i;
typedef __attribute__((address_space(3))) v4i s_v4i;
#else
#define USE_ASYNC_LDS 0
#endif

// ---------------------------------------------------------------------------
// Kernel 1: A[b, i*8+k] = T_k(tanh(x[b,i])) as f16 (one 16B store per thread).
// ---------------------------------------------------------------------------
__global__ __launch_bounds__(256) void kan_basis_f16(const float* __restrict__ x,
                                                     _Float16* __restrict__ A,
                                                     int n) {
  int idx = blockIdx.x * 256 + threadIdx.x;
  if (idx >= n) return;
  float t = tanhf(x[idx]);
  v8h o;
  o[0] = (_Float16)1.0f;
  o[1] = (_Float16)t;
  float a = 1.0f, b = t;
#pragma unroll
  for (int k = 2; k < KCHEB; ++k) {
    float c = 2.0f * t * b - a;
    o[k] = (_Float16)c;
    a = b;
    b = c;
  }
  *(v8h*)(A + (size_t)idx * KCHEB) = o;
}

// ---------------------------------------------------------------------------
// Kernel 2: Wh = (f16)W ; W[o,i,k] row-major is already B^T [1024 x 8192].
// ---------------------------------------------------------------------------
__global__ __launch_bounds__(256) void kan_wconv_f16(const float* __restrict__ W,
                                                     _Float16* __restrict__ Wh,
                                                     int n4) {
  int idx = blockIdx.x * 256 + threadIdx.x;
  if (idx >= n4) return;
  v4f w = *(const v4f*)(W + (size_t)idx * 4);
  v4h h;
  h[0] = (_Float16)w[0];
  h[1] = (_Float16)w[1];
  h[2] = (_Float16)w[2];
  h[3] = (_Float16)w[3];
  *(v4h*)(Wh + (size_t)idx * 4) = h;
}

// ---------------------------------------------------------------------------
// Kernel 3: out = A[4096,8192] * Wh^T + bias.
// 256 threads = 8 waves. Block tile 128x128, wave tile 32x64 (2x4 WMMA tiles).
// K staged through double-buffered LDS in chunks of 64 halves; copies use the
// CDNA5 async global->LDS path (ASYNCcnt) when available, else ds_store.
// ---------------------------------------------------------------------------
__global__ __launch_bounds__(256) void kan_wmma_gemm(const _Float16* __restrict__ A,
                                                     const _Float16* __restrict__ Wh,
                                                     const float* __restrict__ bias,
                                                     float* __restrict__ out) {
  __shared__ _Float16 smemA[2][BM * ASTRIDE];   // 2 x 18KB
  __shared__ _Float16 smemB[2][BN * ASTRIDE];   // 2 x 18KB

  const int tid  = threadIdx.x;
  const int lane = tid & 31;
  const int wave = tid >> 5;
  const int wm   = wave >> 1;          // 0..3  -> 32-row slice of the block
  const int wn   = wave & 1;           // 0..1  -> 64-col slice of the block
  const int lr   = lane & 15;
  const int hi   = (lane >> 4) & 1;    // upper half-wave selector

  const _Float16* gA = A  + (size_t)(blockIdx.y * BM) * KDIM;  // block row base
  const _Float16* gB = Wh + (size_t)(blockIdx.x * BN) * KDIM;  // block col base

  v8f acc[2][4] = {};

#if !USE_ASYNC_LDS
  v8h rA[4], rB[4];   // staging registers for the software-pipelined copy
#endif

  // ---- per-stage cooperative copy: 128 rows x 64 halves = 1024 x 16B chunks,
  //      4 chunks per thread for A and 4 for B. chunk c: row=c>>3, 16B col=c&7.
#if USE_ASYNC_LDS
#define ISSUE_STAGE(k0, buf)                                                    \
  {                                                                             \
    _Pragma("unroll")                                                           \
    for (int i = 0; i < 4; ++i) {                                               \
      int c = tid + i * 256;                                                    \
      int row = c >> 3, c16 = c & 7;                                            \
      const _Float16* gp = gA + (size_t)row * KDIM + (k0) + c16 * 8;            \
      _Float16* sp = &smemA[buf][row * ASTRIDE + c16 * 8];                      \
      __builtin_amdgcn_global_load_async_to_lds_b128((g_v4i*)gp, (s_v4i*)sp, 0, 0); \
      const _Float16* gq = gB + (size_t)row * KDIM + (k0) + c16 * 8;            \
      _Float16* sq = &smemB[buf][row * ASTRIDE + c16 * 8];                      \
      __builtin_amdgcn_global_load_async_to_lds_b128((g_v4i*)gq, (s_v4i*)sq, 0, 0); \
    }                                                                           \
  }
#define COMMIT_STAGE(buf) __builtin_amdgcn_s_wait_asynccnt(0);
#else
#define ISSUE_STAGE(k0, buf)                                                    \
  {                                                                             \
    _Pragma("unroll")                                                           \
    for (int i = 0; i < 4; ++i) {                                               \
      int c = tid + i * 256;                                                    \
      int row = c >> 3, c16 = c & 7;                                            \
      rA[i] = *(const v8h*)(gA + (size_t)row * KDIM + (k0) + c16 * 8);          \
      rB[i] = *(const v8h*)(gB + (size_t)row * KDIM + (k0) + c16 * 8);          \
    }                                                                           \
  }
#define COMMIT_STAGE(buf)                                                       \
  {                                                                             \
    _Pragma("unroll")                                                           \
    for (int i = 0; i < 4; ++i) {                                               \
      int c = tid + i * 256;                                                    \
      int row = c >> 3, c16 = c & 7;                                            \
      *(v8h*)(&smemA[buf][row * ASTRIDE + c16 * 8]) = rA[i];                    \
      *(v8h*)(&smemB[buf][row * ASTRIDE + c16 * 8]) = rB[i];                    \
    }                                                                           \
  }
#endif

  // prologue: fill buffer 0
  ISSUE_STAGE(0, 0)
  COMMIT_STAGE(0)
  __syncthreads();

  const int nst = KDIM / BK;   // 128 stages
  for (int ks = 0; ks < nst; ++ks) {
    const int buf  = ks & 1;
    const int nbuf = buf ^ 1;

    if (ks + 1 < nst) ISSUE_STAGE((ks + 1) * BK, nbuf)

    // ---- compute this stage: 2 K-steps of 32, 8 WMMAs each
    const _Float16* As = smemA[buf];
    const _Float16* Bs = smemB[buf];
#pragma unroll
    for (int kk = 0; kk < BK; kk += 32) {
      v16h af[2];
#pragma unroll
      for (int t = 0; t < 2; ++t) {
        // A frag: lane holds row (lr); K chunks kb..kb+7 and kb+16..kb+23,
        // kb = kk + (hi ? 8 : 0)
        int row = wm * 32 + t * 16 + lr;
        int kb  = kk + (hi ? 8 : 0);
        const _Float16* p = As + row * ASTRIDE + kb;
        v8h lo = *(const v8h*)(p);
        v8h h8 = *(const v8h*)(p + 16);
#pragma unroll
        for (int e = 0; e < 8; ++e) { af[t][e] = lo[e]; af[t][e + 8] = h8[e]; }
      }
      v16h bf[4];
#pragma unroll
      for (int j = 0; j < 4; ++j) {
        // B frag: lane holds col (lr); K chunks kb..kb+7 and kb+16..kb+23, kb = kk + (hi?8:0)
        int col = wn * 64 + j * 16 + lr;
        int kb  = kk + (hi ? 8 : 0);
        const _Float16* p = Bs + col * ASTRIDE + kb;
        v8h lo = *(const v8h*)(p);
        v8h h8 = *(const v8h*)(p + 16);
#pragma unroll
        for (int e = 0; e < 8; ++e) { bf[j][e] = lo[e]; bf[j][e + 8] = h8[e]; }
      }
#pragma unroll
      for (int t = 0; t < 2; ++t)
#pragma unroll
        for (int j = 0; j < 4; ++j)
          acc[t][j] = __builtin_amdgcn_wmma_f32_16x16x32_f16(
              false, af[t], false, bf[j], (short)0, acc[t][j], false, false);
    }

    if (ks + 1 < nst) COMMIT_STAGE(nbuf)
    __syncthreads();
  }

  // ---- epilogue: C/D layout lanes 0-15 -> M=r, lanes 16-31 -> M=r+8; N=lr.
#pragma unroll
  for (int j = 0; j < 4; ++j) {
    int n = blockIdx.x * BN + wn * 64 + j * 16 + lr;
    float bs = bias[n];
#pragma unroll
    for (int t = 0; t < 2; ++t) {
      int mBase = blockIdx.y * BM + wm * 32 + t * 16 + hi * 8;
#pragma unroll
      for (int r = 0; r < 8; ++r) {
        out[(size_t)(mBase + r) * OUTF + n] = acc[t][j][r] + bs;
      }
    }
  }
#undef ISSUE_STAGE
#undef COMMIT_STAGE
}

// ---------------------------------------------------------------------------
extern "C" void kernel_launch(void* const* d_in, const int* in_sizes, int n_in,
                              void* d_out, int out_size, void* d_ws, size_t ws_size,
                              hipStream_t stream) {
  (void)in_sizes; (void)n_in; (void)out_size; (void)ws_size;

  const float* x    = (const float*)d_in[0];  // [4096, 1024]
  const float* W    = (const float*)d_in[1];  // [1024, 1024, 8]
  const float* bias = (const float*)d_in[2];  // [1024]
  float* out        = (float*)d_out;          // [4096, 1024]

  _Float16* A  = (_Float16*)d_ws;                                      // 64 MB
  _Float16* Wh = (_Float16*)((char*)d_ws + (size_t)BATCH * KDIM * 2);  // +16 MB

  int nA = BATCH * INF;
  kan_basis_f16<<<dim3((nA + 255) / 256), dim3(256), 0, stream>>>(x, A, nA);

  int n4 = (OUTF * INF * KCHEB) / 4;
  kan_wconv_f16<<<dim3((n4 + 255) / 256), dim3(256), 0, stream>>>(W, Wh, n4);

  dim3 grid(OUTF / BN, BATCH / BM);   // (8, 32)
  kan_wmma_gemm<<<grid, dim3(256), 0, stream>>>(A, Wh, bias, out);
}
